// GCN_ssm_block_18648747999525
// MI455X (gfx1250) — matrix operand order repaired and checked by reading, not compiled, not run on hardware
//
#include <hip/hip_runtime.h>

// ---------------- CDNA5 types ----------------
typedef __bf16 bf16;
typedef __attribute__((ext_vector_type(16))) __bf16 v16bf;
typedef __attribute__((ext_vector_type(8)))  __bf16 v8bf;
typedef __attribute__((ext_vector_type(8)))  float  v8f;
typedef __attribute__((ext_vector_type(4)))  unsigned int u32x4;
typedef __attribute__((ext_vector_type(8)))  int i32x8;
typedef __attribute__((ext_vector_type(4)))  int i32x4;

#define N_NODES 8192
#define N_EDGES 65536
#define D_MODEL 64
#define D_INNER 128
#define D_STATE 16
#define DT_RANK 4
#define N_LAYER 3
#define STATE_F (D_INNER * D_STATE)   // 2048 floats per node (8 KB)
#define XDBL_LD 48                    // 36 padded to 48 (3 WMMA n-tiles)

#if defined(__HIP_DEVICE_COMPILE__) && __has_builtin(__builtin_amdgcn_tensor_load_to_lds)
#define HAVE_TDM 1
#else
#define HAVE_TDM 0
#endif

__device__ __forceinline__ bf16 f2bf(float f) {
    union { float f; unsigned u; } v; v.f = f;
    unsigned r = (v.u + 0x7FFFu + ((v.u >> 16) & 1u)) >> 16;
    unsigned short h = (unsigned short)r;
    return __builtin_bit_cast(bf16, h);
}

// ---------------- WMMA fragment helpers ----------------
// A frag (16-bit A 16x32, ISA 05_wmma): lane L -> row M=L%16, kbase=(L<16?0:8);
// half h -> K = kbase + (h>=8?16:0) + (h&7)  ==> two contiguous 16B runs per lane.
__device__ __forceinline__ v16bf load_a_frag(const bf16* __restrict__ A, int lda, int k0) {
    const int lane = threadIdx.x & 31;
    const int m = lane & 15;
    const int ka = k0 + ((lane >> 4) << 3);
    const v8bf lo = *(const v8bf*)(A + m * lda + ka);
    const v8bf hi = *(const v8bf*)(A + m * lda + ka + 16);
    return __builtin_shufflevector(lo, hi, 0, 1, 2, 3, 4, 5, 6, 7,
                                   8, 9, 10, 11, 12, 13, 14, 15);
}

// B frags come from weights pre-packed into exact lane layout:
// P[((ntile*KC + kc)*32 + lane)*16 + h] = W[(kc*32 + (lane>>4)*8 + (h>=8?16:0) + (h&7))][ntile*16 + (lane&15)]
template <int K>
__device__ __forceinline__ v8f wmma_tile_packed(const bf16* __restrict__ A, int lda,
                                                const bf16* __restrict__ Bp, int ntile,
                                                v8f acc) {
    const int lane = threadIdx.x & 31;
    constexpr int KC = K / 32;
#pragma unroll
    for (int kc = 0; kc < KC; ++kc) {
        v16bf a = load_a_frag(A, lda, kc * 32);
        const v16bf b = *(const v16bf*)(Bp + (((size_t)(ntile * KC + kc) * 32 + lane) << 4));
        acc = __builtin_amdgcn_wmma_f32_16x16x32_bf16(false, a, false, b,
                                                      (short)0, acc, false, false);
    }
    return acc;
}

// C/D frag: VGPR r -> row M = r + (lane<16?0:8), col N = lane&15.
__device__ __forceinline__ void store_tile_f32(float* __restrict__ C, int ldc, v8f acc) {
    const int lane = threadIdx.x & 31;
    const int n  = lane & 15;
    const int mo = (lane >> 4) * 8;
#pragma unroll
    for (int r = 0; r < 8; ++r) C[(mo + r) * ldc + n] = acc[r];
}

// ---------------- weight packing (float -> bf16, WMMA-frag layout) ---------
__global__ void k_pack_wb(const float* __restrict__ W, bf16* __restrict__ P,
                          int K, int N, int srcN) {
    int i = blockIdx.x * blockDim.x + threadIdx.x;   // over K*N packed elements
    if (i >= K * N) return;
    const int KC = K / 32;
    int h = i & 15;
    int lane = (i >> 4) & 31;
    int t = i >> 9;
    int ntile = t / KC, kc = t % KC;
    int k = kc * 32 + ((lane >> 4) << 3) + ((h >> 3) << 4) + (h & 7);
    int n = ntile * 16 + (lane & 15);
    float v = (n < srcN) ? W[k * srcN + n] : 0.0f;
    P[i] = f2bf(v);
}

__global__ void k_x_to_bf16(const float* __restrict__ x, bf16* __restrict__ xb) {
    int i = blockIdx.x * blockDim.x + threadIdx.x;
    if (i < N_NODES * D_MODEL) xb[i] = f2bf(x[i]);
}

// ---------------- GEMM 1: x(8192x64) @ in_proj(64x256), fused relu/split ---
__global__ void k_gemm_in(const bf16* __restrict__ xb, const bf16* __restrict__ wp,
                          float* __restrict__ xs, float* __restrict__ res,
                          bf16* __restrict__ xs_b) {
    const int wid = (blockIdx.x * blockDim.x + threadIdx.x) >> 5;   // 8192 waves exactly
    const int NT = (2 * D_INNER) / 16;                              // 16 n-tiles
    const int tm = wid / NT, tn = wid % NT;
    v8f acc = {};
    acc = wmma_tile_packed<D_MODEL>(xb + tm * 16 * D_MODEL, D_MODEL, wp, tn, acc);
    const int lane = threadIdx.x & 31;
    const int nc = tn * 16 + (lane & 15);
    const int mo = tm * 16 + (lane >> 4) * 8;
    if (nc < D_INNER) {                       // xs half: relu + fp32 + bf16 copies
#pragma unroll
        for (int r = 0; r < 8; ++r) {
            float v = fmaxf(acc[r], 0.0f);
            xs  [(mo + r) * D_INNER + nc] = v;
            xs_b[(mo + r) * D_INNER + nc] = f2bf(v);
        }
    } else {                                  // res half
#pragma unroll
        for (int r = 0; r < 8; ++r) res[(mo + r) * D_INNER + (nc - D_INNER)] = acc[r];
    }
}

// ---------------- GEMM 2: xs(8192x128) @ x_proj(128x48pad) -> x_dbl --------
__global__ void k_gemm_xdbl(const bf16* __restrict__ xs_b, const bf16* __restrict__ wp,
                            float* __restrict__ xdbl) {
    const int wid = (blockIdx.x * blockDim.x + threadIdx.x) >> 5;   // 1536 waves exactly
    const int NT = XDBL_LD / 16;                                    // 3
    const int tm = wid / NT, tn = wid % NT;
    v8f acc = {};
    acc = wmma_tile_packed<D_INNER>(xs_b + tm * 16 * D_INNER, D_INNER, wp, tn, acc);
    store_tile_f32(xdbl + tm * 16 * XDBL_LD + tn * 16, XDBL_LD, acc);
}

// ---------------- per-(node,channel): delta = softplus(dt @ dt_proj) -------
__global__ void k_node_prep(const float* __restrict__ xdbl, const float* __restrict__ dtw,
                            const float* __restrict__ xs,
                            float* __restrict__ delta, float* __restrict__ du) {
    int t = blockIdx.x * blockDim.x + threadIdx.x;           // (b,d)
    if (t >= N_NODES * D_INNER) return;
    int b = t >> 7, d = t & (D_INNER - 1);
    float p = 0.f;
#pragma unroll
    for (int r = 0; r < DT_RANK; ++r) p += xdbl[b * XDBL_LD + r] * dtw[r * D_INNER + d];
    float sp = (p > 20.f) ? p : log1pf(expf(p));
    delta[t] = sp;
    du[t]    = sp * xs[t];
}

// ---------------- graph normalization ----------------
__global__ void k_edges_ones(float* __restrict__ w) {
    int e = blockIdx.x * blockDim.x + threadIdx.x;
    if (e < N_EDGES) w[e] = 1.0f;
}
__global__ void k_deg_zero(float* __restrict__ deg) {
    int b = blockIdx.x * blockDim.x + threadIdx.x;
    if (b < N_NODES) deg[b] = 0.0f;
}
__global__ void k_deg_accum(const int* __restrict__ ei, const float* __restrict__ w,
                            float* __restrict__ deg) {
    int e = blockIdx.x * blockDim.x + threadIdx.x;
    if (e < N_EDGES) atomicAdd(&deg[ei[e]], w[e]);
}
__global__ void k_dinv(const float* __restrict__ deg, float* __restrict__ dinv,
                       float* __restrict__ dw) {
    int b = blockIdx.x * blockDim.x + threadIdx.x;
    if (b < N_NODES) {
        float di = rsqrtf(deg[b] + 1.0f);     // deg+1 > 0 always
        dinv[b] = di;
        dw[b]   = di * di;
    }
}
__global__ void k_ew(const int* __restrict__ ei, const float* __restrict__ w,
                     const float* __restrict__ dinv, float* __restrict__ ew) {
    int e = blockIdx.x * blockDim.x + threadIdx.x;
    if (e < N_EDGES) ew[e] = dinv[ei[e]] * w[e] * dinv[ei[N_EDGES + e]];
}

// ---------------- SSM recurrence + y reduction (float4 path) ---------------
__global__ void k_state_update(const float* __restrict__ delta, const float* __restrict__ du,
                               const float* __restrict__ A_log, const float* __restrict__ xdbl,
                               float* __restrict__ state, float* __restrict__ y, int first) {
    int t = blockIdx.x * blockDim.x + threadIdx.x;           // (b,d)
    if (t >= N_NODES * D_INNER) return;
    int b = t >> 7, d = t & (D_INNER - 1);
    const float de  = delta[t];
    const float duu = du[t];
    float al[D_STATE], Bl[D_STATE], Cl[D_STATE], sl[D_STATE];
#pragma unroll
    for (int q = 0; q < 4; ++q) {
        ((float4*)al)[q] = ((const float4*)(A_log + d * D_STATE))[q];
        ((float4*)Bl)[q] = ((const float4*)(xdbl + b * XDBL_LD + DT_RANK))[q];
        ((float4*)Cl)[q] = ((const float4*)(xdbl + b * XDBL_LD + DT_RANK + D_STATE))[q];
        if (!first) ((float4*)sl)[q] = ((const float4*)(state + (size_t)t * D_STATE))[q];
    }
    float acc = 0.f;
#pragma unroll
    for (int n = 0; n < D_STATE; ++n) {
        float s0 = first ? 0.0f : sl[n];
        float sn = expf(de * -expf(al[n])) * s0 + duu * Bl[n];
        sl[n] = sn;
        acc  += sn * Cl[n];
    }
#pragma unroll
    for (int q = 0; q < 4; ++q)
        ((float4*)(state + (size_t)t * D_STATE))[q] = ((float4*)sl)[q];
    y[t] = acc;
}

// ---------------- SpMM: out = dw*state + scatter(ew * state[col]) ----------
__global__ void k_spmm_self(const float* __restrict__ sin_, const float* __restrict__ dw,
                            float* __restrict__ sout) {
    int t = blockIdx.x * blockDim.x + threadIdx.x;           // float4 index
    if (t >= N_NODES * (STATE_F / 4)) return;
    int b = t >> 9;                                          // 512 float4 per node
    float4 v = ((const float4*)sin_)[t];
    float w = dw[b];
    ((float4*)sout)[t] = make_float4(w * v.x, w * v.y, w * v.z, w * v.w);
}

#if HAVE_TDM
// Build a 2D Tensor-DMA descriptor (D# group0/group1, ISA 08_async_tensor 8.3/8.4)
// copying one 2048-float row from global to LDS, and issue TENSOR_LOAD_TO_LDS.
__device__ __forceinline__ void tdm_load_row(unsigned lds_off, const float* __restrict__ src) {
    unsigned long long ga = (unsigned long long)(size_t)src;
    u32x4 g0;
    g0.x = 1u;                                            // count=1 (valid), user mode
    g0.y = lds_off;                                       // lds_addr [63:32]
    g0.z = (unsigned)(ga & 0xFFFFFFFFull);                // global_addr [95:64]
    g0.w = (unsigned)((ga >> 32) & 0x01FFFFFFull) | (2u << 30);  // addr[56:32] | type=2
    i32x8 g1;
    g1[0] = (int)(2u << 16);                              // wg_mask=0, data_size=4B
    g1[1] = (int)((2048u & 0xFFFFu) << 16);               // tensor_dim0[15:0] @ bit48
    g1[2] = (int)((2048u >> 16) | (1u << 16));            // dim0[31:16] | tensor_dim1=1
    g1[3] = (int)(2048u << 16);                           // dim1 hi | tile_dim0=2048
    g1[4] = 1;                                            // tile_dim1=1, tile_dim2=0
    g1[5] = 2048;                                         // tensor_dim0_stride (low 32)
    g1[6] = 0;
    g1[7] = 0;
    i32x4 z4 = {0, 0, 0, 0};
#if defined(__clang_major__) && (__clang_major__ >= 23)
    i32x8 z8 = {0, 0, 0, 0, 0, 0, 0, 0};
    __builtin_amdgcn_tensor_load_to_lds(g0, g1, z4, z4, z8, 0);
#else
    __builtin_amdgcn_tensor_load_to_lds(g0, g1, z4, z4, 0);
#endif
}
#endif

__global__ void k_spmm_edges(const int* __restrict__ ei, const float* __restrict__ ew,
                             const float* __restrict__ sin_, float* __restrict__ sout) {
    const int e = blockIdx.x;
    const float w = ew[e];
    if (w == 0.0f) return;                                   // gated-out edge (block-uniform)
    const int r = ei[e], c = ei[N_EDGES + e];
    float* __restrict__ dst = sout + (size_t)r * STATE_F;
#if HAVE_TDM
    __shared__ __align__(16) float lrow[STATE_F];
    if (threadIdx.x < 32) {                                  // wave 0 drives the TDM
        tdm_load_row((unsigned)(size_t)(void*)lrow, sin_ + (size_t)c * STATE_F);
#if __has_builtin(__builtin_amdgcn_s_wait_tensorcnt)
        __builtin_amdgcn_s_wait_tensorcnt(0);
#endif
    }
    __syncthreads();
#pragma unroll
    for (int j = 0; j < 8; ++j) {                            // 256 thr * 8 = 2048 floats
        int ci = threadIdx.x + j * 256;
        atomicAdd(dst + ci, w * lrow[ci]);
    }
#else
    const float4* __restrict__ src = (const float4*)(sin_ + (size_t)c * STATE_F);
#pragma unroll
    for (int j = 0; j < 2; ++j) {
        int ci = threadIdx.x * 2 + j;
        float4 v = src[ci];
        atomicAdd(dst + ci * 4 + 0, w * v.x);
        atomicAdd(dst + ci * 4 + 1, w * v.y);
        atomicAdd(dst + ci * 4 + 2, w * v.z);
        atomicAdd(dst + ci * 4 + 3, w * v.w);
    }
#endif
}

// ---------------- Gumbel-hard gating (forward == argmax one-hot) -----------
__device__ __forceinline__ float gumbel_noise(unsigned layer, unsigned node,
                                              unsigned which, unsigned cls) {
    unsigned long long z = ((unsigned long long)layer << 40) ^
                           ((unsigned long long)node  << 16) ^
                           ((unsigned long long)which << 8) ^ cls;
    z += 0x9E3779B97F4A7C15ull;
    z = (z ^ (z >> 30)) * 0xBF58476D1CE4E5B9ull;
    z = (z ^ (z >> 27)) * 0x94D049BB133111EBull;
    z ^= z >> 31;
    float u = (float)((z >> 40) + 1ull) * (1.0f / 16777217.0f);   // (0,1]
    return -logf(-logf(u));
}

__global__ void k_gating(const float* __restrict__ y,
                         const float* __restrict__ inw, const float* __restrict__ inb,
                         const float* __restrict__ outw, const float* __restrict__ outb,
                         float* __restrict__ inh, float* __restrict__ outh, int layer) {
    const int b    = (blockIdx.x * blockDim.x + threadIdx.x) >> 5;  // one wave per node
    const int lane = threadIdx.x & 31;
    if (b >= N_NODES) return;
    float s0 = 0.f, s1 = 0.f, s2 = 0.f, s3 = 0.f;
#pragma unroll
    for (int t = 0; t < 4; ++t) {
        int d = lane * 4 + t;
        float yy = y[b * D_INNER + d];
        s0 += yy * inw [d * 2 + 0];
        s1 += yy * inw [d * 2 + 1];
        s2 += yy * outw[d * 2 + 0];
        s3 += yy * outw[d * 2 + 1];
    }
#pragma unroll
    for (int off = 16; off > 0; off >>= 1) {
        s0 += __shfl_down(s0, off);
        s1 += __shfl_down(s1, off);
        s2 += __shfl_down(s2, off);
        s3 += __shfl_down(s3, off);
    }
    if (lane == 0) {
        float li0 = s0 + inb [0] + gumbel_noise(layer, b, 0, 0);
        float li1 = s1 + inb [1] + gumbel_noise(layer, b, 0, 1);
        float lo0 = s2 + outb[0] + gumbel_noise(layer, b, 1, 0);
        float lo1 = s3 + outb[1] + gumbel_noise(layer, b, 1, 1);
        inh [b] = (li0 >= li1) ? 1.0f : 0.0f;   // one_hot(argmax)[0]
        outh[b] = (lo0 >= lo1) ? 1.0f : 0.0f;
    }
}

__global__ void k_reweight(const int* __restrict__ ei, const float* __restrict__ inh,
                           const float* __restrict__ outh, float* __restrict__ w) {
    int e = blockIdx.x * blockDim.x + threadIdx.x;
    if (e < N_EDGES) w[e] = inh[ei[N_EDGES + e]] * outh[ei[e]];   // in_p[col]*out_p[row]
}

// ---------------- finalize + out projection ----------------
__global__ void k_finalize(const float* __restrict__ y, const float* __restrict__ xs,
                           const float* __restrict__ res, const float* __restrict__ Dv,
                           bf16* __restrict__ pre_b) {
    int t = blockIdx.x * blockDim.x + threadIdx.x;
    if (t >= N_NODES * D_INNER) return;
    int d = t & (D_INNER - 1);
    float v = (y[t] + xs[t] * Dv[d]) * fmaxf(res[t], 0.0f);
    pre_b[t] = f2bf(v);
}

__global__ void k_gemm_out(const bf16* __restrict__ pre_b, const bf16* __restrict__ wp,
                           float* __restrict__ out) {
    const int wid = (blockIdx.x * blockDim.x + threadIdx.x) >> 5;   // 2048 waves exactly
    const int NT = D_MODEL / 16;                                    // 4
    const int tm = wid / NT, tn = wid % NT;
    v8f acc = {};
    acc = wmma_tile_packed<D_INNER>(pre_b + tm * 16 * D_INNER, D_INNER, wp, tn, acc);
    store_tile_f32(out + tm * 16 * D_MODEL + tn * 16, D_MODEL, acc);
}

// ---------------- host launch ----------------
extern "C" void kernel_launch(void* const* d_in, const int* in_sizes, int n_in,
                              void* d_out, int out_size, void* d_ws, size_t ws_size,
                              hipStream_t stream) {
    const float* x      = (const float*)d_in[0];
    const float* win    = (const float*)d_in[1];
    const float* wxp    = (const float*)d_in[2];
    const float* dtw    = (const float*)d_in[3];
    const float* A_log  = (const float*)d_in[4];
    const float* Dv     = (const float*)d_in[5];
    const float* wout   = (const float*)d_in[6];
    const float* inw    = (const float*)d_in[7];
    const float* inb    = (const float*)d_in[8];
    const float* outw   = (const float*)d_in[9];
    const float* outb   = (const float*)d_in[10];
    const int*   ei     = (const int*)d_in[11];   // [2, N_EDGES] (x64-disabled jax -> i32)
    float*       out    = (float*)d_out;

    char* p = (char*)d_ws;
    auto alloc = [&](size_t bytes) -> char* {
        char* r = p; p += (bytes + 255) & ~(size_t)255; return r;
    };
    float* state0 = (float*)alloc((size_t)N_NODES * STATE_F * 4);
    float* state1 = (float*)alloc((size_t)N_NODES * STATE_F * 4);
    float* xs     = (float*)alloc((size_t)N_NODES * D_INNER * 4);
    float* res    = (float*)alloc((size_t)N_NODES * D_INNER * 4);
    float* delta  = (float*)alloc((size_t)N_NODES * D_INNER * 4);
    float* du     = (float*)alloc((size_t)N_NODES * D_INNER * 4);
    float* y      = (float*)alloc((size_t)N_NODES * D_INNER * 4);
    float* xdbl   = (float*)alloc((size_t)N_NODES * XDBL_LD * 4);
    float* wedge  = (float*)alloc((size_t)N_EDGES * 4);
    float* ew     = (float*)alloc((size_t)N_EDGES * 4);
    float* deg    = (float*)alloc((size_t)N_NODES * 4);
    float* dinv   = (float*)alloc((size_t)N_NODES * 4);
    float* dw     = (float*)alloc((size_t)N_NODES * 4);
    float* inh    = (float*)alloc((size_t)N_NODES * 4);
    float* outh   = (float*)alloc((size_t)N_NODES * 4);
    bf16* xb      = (bf16*)alloc((size_t)N_NODES * D_MODEL * 2);
    bf16* xs_b    = (bf16*)alloc((size_t)N_NODES * D_INNER * 2);
    bf16* pre_b   = (bf16*)alloc((size_t)N_NODES * D_INNER * 2);
    bf16* win_p   = (bf16*)alloc((size_t)D_MODEL * 2 * D_INNER * 2);
    bf16* wxp_p   = (bf16*)alloc((size_t)D_INNER * XDBL_LD * 2);
    bf16* wout_p  = (bf16*)alloc((size_t)D_INNER * D_MODEL * 2);

    const int T = 256;
    // pack weights into WMMA fragment layout (bf16)
    k_pack_wb<<<(D_MODEL * 2 * D_INNER + T - 1) / T, T, 0, stream>>>(win, win_p,
                                                                     D_MODEL, 2 * D_INNER,
                                                                     2 * D_INNER);
    k_pack_wb<<<(D_INNER * XDBL_LD + T - 1) / T, T, 0, stream>>>(wxp, wxp_p,
                                                                 D_INNER, XDBL_LD, 36);
    k_pack_wb<<<(D_INNER * D_MODEL + T - 1) / T, T, 0, stream>>>(wout, wout_p,
                                                                 D_INNER, D_MODEL, D_MODEL);
    k_x_to_bf16<<<(N_NODES * D_MODEL + T - 1) / T, T, 0, stream>>>(x, xb);

    // projections (once: xl is a broadcast of x across layers)
    k_gemm_in<<<(N_NODES / 16) * 16 / 8, T, 0, stream>>>(xb, win_p, xs, res, xs_b);
    k_gemm_xdbl<<<(N_NODES / 16) * (XDBL_LD / 16) / 8, T, 0, stream>>>(xs_b, wxp_p, xdbl);
    k_node_prep<<<(N_NODES * D_INNER + T - 1) / T, T, 0, stream>>>(xdbl, dtw, xs, delta, du);

    // initial normalized adjacency (w = 1)
    k_edges_ones<<<N_EDGES / T, T, 0, stream>>>(wedge);
    k_deg_zero<<<N_NODES / T, T, 0, stream>>>(deg);
    k_deg_accum<<<N_EDGES / T, T, 0, stream>>>(ei, wedge, deg);
    k_dinv<<<N_NODES / T, T, 0, stream>>>(deg, dinv, dw);
    k_ew<<<N_EDGES / T, T, 0, stream>>>(ei, wedge, dinv, ew);

    float* cur = state0;
    float* nxt = state1;
    for (int i = 0; i < N_LAYER; ++i) {
        k_state_update<<<(N_NODES * D_INNER + T - 1) / T, T, 0, stream>>>(
            delta, du, A_log, xdbl, cur, y, i == 0);
        k_spmm_self<<<(N_NODES * (STATE_F / 4) + T - 1) / T, T, 0, stream>>>(cur, dw, nxt);
        k_spmm_edges<<<N_EDGES, T, 0, stream>>>(ei, ew, cur, nxt);
        { float* t = cur; cur = nxt; nxt = t; }
        k_gating<<<(N_NODES * 32 + T - 1) / T, T, 0, stream>>>(
            y, inw, inb, outw, outb, inh, outh, i);
        k_reweight<<<N_EDGES / T, T, 0, stream>>>(ei, inh, outh, wedge);
        k_deg_zero<<<N_NODES / T, T, 0, stream>>>(deg);
        k_deg_accum<<<N_EDGES / T, T, 0, stream>>>(ei, wedge, deg);
        k_dinv<<<N_NODES / T, T, 0, stream>>>(deg, dinv, dw);
        k_ew<<<N_EDGES / T, T, 0, stream>>>(ei, wedge, dinv, ew);
    }

    // out = ((y_last + xs*D) * relu(res)) @ out_proj  (only layer -1 is emitted)
    k_finalize<<<(N_NODES * D_INNER + T - 1) / T, T, 0, stream>>>(y, xs, res, Dv, pre_b);
    k_gemm_out<<<(N_NODES / 16) * (D_MODEL / 16) / 8, T, 0, stream>>>(pre_b, wout_p, out);
    (void)in_sizes; (void)n_in; (void)out_size; (void)ws_size;
}
